// MultiHeadAttentionEfficient_62912680952506
// MI455X (gfx1250) — matrix operand order repaired and checked by reading, not compiled, not run on hardware
//
#include <hip/hip_runtime.h>
#include <hip/hip_bf16.h>
#include <stdint.h>

// Problem constants (fixed by the reference)
#define B_SZ 4
#define T_SZ 2048
#define C_SZ 1024
#define H_SZ 16
#define HS_SZ 64
#define M_SZ (B_SZ * T_SZ)   // 8192 rows
#define N_QKV (3 * C_SZ)     // 3072
#define K_SZ C_SZ            // 1024

typedef __attribute__((ext_vector_type(16))) __bf16 v16bf;
typedef __attribute__((ext_vector_type(8)))  float  v8f;
typedef __attribute__((ext_vector_type(4)))  unsigned int v4u;
typedef __attribute__((ext_vector_type(8)))  unsigned int v8u;

union Frag {
    v16bf v;
    unsigned int u[8];
};

__device__ __forceinline__ v8f wmma_bf16(const v16bf& a, const v16bf& b, const v8f& c) {
    // (neg_a, A, neg_b, B, c_mod, C, reuse_a, reuse_b)
    return __builtin_amdgcn_wmma_f32_16x16x32_bf16(false, a, false, b, (short)0, c, false, false);
}

__device__ __forceinline__ unsigned short f2bf(float f) {
    unsigned int u = __float_as_uint(f);
    u += 0x7fffu + ((u >> 16) & 1u);   // round-to-nearest-even
    return (unsigned short)(u >> 16);
}

// ---------------------------------------------------------------------------
// TDM: issue a 64x64 bf16 2D tile load (row stride 64 elements) into LDS.
// D# layout per CDNA5 ISA:
//   group0: count=1 | lds_addr | global_addr[56:0] | type=2
//   group1: wg_mask=0, data_size=1(2B), tensor_dim0=64, tensor_dim1=2048,
//           tile_dim0=64, tile_dim1=64, tensor_dim0_stride=64
// Tracked on TENSORcnt; caller must s_wait_tensorcnt before consuming.
// ---------------------------------------------------------------------------
__device__ __forceinline__ void tdm_load_tile64x64_bf16(unsigned lds_addr,
                                                        const unsigned short* gptr) {
    unsigned long long ga = (unsigned long long)(uintptr_t)gptr;
    v4u g0;
    g0[0] = 1u;                                                   // count=1
    g0[1] = lds_addr;                                             // LDS byte offset
    g0[2] = (unsigned)ga;                                         // global addr lo
    g0[3] = ((unsigned)(ga >> 32) & 0x01FFFFFFu) | 0x80000000u;   // addr hi | type=2
    v8u g1;
    g1[0] = 0x00010000u;   // workgroup_mask=0 (not clustered), data_size=2B
    g1[1] = 64u << 16;     // tensor_dim0 = 64
    g1[2] = 2048u << 16;   // tensor_dim1 = 2048
    g1[3] = 64u << 16;     // tile_dim0 = 64
    g1[4] = 64u;           // tile_dim1 = 64, tile_dim2 = 0
    g1[5] = 64u;           // tensor_dim0_stride = 64 (elements)
    g1[6] = 0u;
    g1[7] = 0u;
    asm volatile("tensor_load_to_lds %0, %1" :: "s"(g0), "s"(g1) : "memory");
}

// ---------------------------------------------------------------------------
// fp32 -> bf16 conversion (grid-stride)
// ---------------------------------------------------------------------------
__global__ void cvt_f32_bf16(const float* __restrict__ src,
                             unsigned short* __restrict__ dst, int n) {
    int stride = gridDim.x * blockDim.x;
    for (int i = blockIdx.x * blockDim.x + threadIdx.x; i < n; i += stride)
        dst[i] = f2bf(src[i]);
}

// ---------------------------------------------------------------------------
// QKV GEMM: [8192,1024] x [1024,3072] -> q/k/v in [B,H,T,HS] bf16 layout.
// Block tile 128x128, 8 waves, wave tile 64x32 (4x2 WMMA tiles), BK=32.
// Q is pre-scaled by 1/sqrt(HS)=0.125 at the epilogue.
// ---------------------------------------------------------------------------
__global__ __launch_bounds__(256) void qkv_gemm(
    const unsigned short* __restrict__ A,    // x, bf16 [M,K]
    const unsigned short* __restrict__ Bm,   // W_qkv, bf16 [K,N]
    unsigned short* __restrict__ qh,
    unsigned short* __restrict__ kh,
    unsigned short* __restrict__ vh) {
    __shared__ unsigned short As[128][32];   // [m][k]
    __shared__ unsigned short Bs[128][32];   // [n][k] (transposed for dword gathers)

    const int tid   = threadIdx.x;
    const int wave  = tid >> 5;
    const int lane  = tid & 31;
    const int mlan  = lane & 15;
    const int hlan  = lane >> 4;
    const int waveM = wave >> 2;             // 0..1 -> 64 rows
    const int waveN = wave & 3;              // 0..3 -> 32 cols

    const int rowBlk = blockIdx.y * 128;
    const int colBlk = blockIdx.x * 128;

    v8f acc[4][2] = {};

    const int arow = tid >> 2;               // 0..63
    const int acol = (tid & 3) * 8;          // 0..24
    const int brow = tid >> 4;               // 0..15
    const int bcol = (tid & 15) * 8;         // 0..120

    for (int k0 = 0; k0 < K_SZ; k0 += 32) {
        __syncthreads();
#pragma unroll
        for (int p = 0; p < 2; ++p) {
            int r = p * 64 + arow;
            *(uint4*)&As[r][acol] =
                *(const uint4*)&A[(size_t)(rowBlk + r) * K_SZ + k0 + acol];
        }
#pragma unroll
        for (int p = 0; p < 2; ++p) {
            int kk = p * 16 + brow;
            uint4 t = *(const uint4*)&Bm[(size_t)(k0 + kk) * N_QKV + colBlk + bcol];
            const unsigned short* e = (const unsigned short*)&t;
#pragma unroll
            for (int i = 0; i < 8; ++i) Bs[bcol + i][kk] = e[i];
        }
        if (k0 + 32 < K_SZ) {
            __builtin_prefetch(&A[(size_t)(rowBlk + arow) * K_SZ + k0 + 32 + acol], 0, 3);
            __builtin_prefetch(&Bm[(size_t)(k0 + 32 + brow) * N_QKV + colBlk + bcol], 0, 3);
        }
        __syncthreads();

        Frag af[4];
#pragma unroll
        for (int mt = 0; mt < 4; ++mt) {
            int rr = waveM * 64 + mt * 16 + mlan;
#pragma unroll
            for (int v = 0; v < 8; ++v) {
                int kk = ((v >> 2) << 4) + hlan * 8 + ((v & 3) << 1);
                af[mt].u[v] = *(const unsigned int*)&As[rr][kk];
            }
        }
        Frag bf2[2];
#pragma unroll
        for (int nt = 0; nt < 2; ++nt) {
            int cc = waveN * 32 + nt * 16 + mlan;
#pragma unroll
            for (int v = 0; v < 8; ++v) {
                int kk = ((v >> 2) << 4) + hlan * 8 + ((v & 3) << 1);
                bf2[nt].u[v] = *(const unsigned int*)&Bs[cc][kk];
            }
        }
#pragma unroll
        for (int mt = 0; mt < 4; ++mt)
#pragma unroll
            for (int nt = 0; nt < 2; ++nt)
                acc[mt][nt] = wmma_bf16(af[mt].v, bf2[nt].v, acc[mt][nt]);
    }

#pragma unroll
    for (int mt = 0; mt < 4; ++mt) {
#pragma unroll
        for (int nt = 0; nt < 2; ++nt) {
#pragma unroll
            for (int r = 0; r < 8; ++r) {
                int row = rowBlk + waveM * 64 + mt * 16 + r + 8 * hlan;
                int col = colBlk + waveN * 32 + nt * 16 + mlan;
                float vv = acc[mt][nt][r];
                int which = col >> 10;
                int c = col & (C_SZ - 1);
                int head = c >> 6, d = c & 63;
                if (which == 0) vv *= 0.125f;  // fold 1/sqrt(HS) into Q
                size_t dst =
                    (((size_t)((row >> 11) * H_SZ + head)) * T_SZ + (row & (T_SZ - 1))) * HS_SZ + d;
                unsigned short bb = f2bf(vv);
                if (which == 0) qh[dst] = bb;
                else if (which == 1) kh[dst] = bb;
                else vh[dst] = bb;
            }
        }
    }
}

// ---------------------------------------------------------------------------
// Flash attention: one block = one (b,h) x 128 query rows; 8 waves, 16 q rows
// each. K/V 64x64 bf16 tiles staged by the Tensor Data Mover into
// double-buffered LDS (DMA of tile kt+1 overlaps WMMA on tile kt), online
// softmax with wave32 shfl_xor row reductions, P re-laid-out via per-wave LDS.
// ---------------------------------------------------------------------------
__global__ __launch_bounds__(256) void flash_attn(
    const unsigned short* __restrict__ qh,
    const unsigned short* __restrict__ kh,
    const unsigned short* __restrict__ vh,
    unsigned short* __restrict__ attnb) {
    __shared__ unsigned short Ks[2][64][64];     // [buf][key][d]
    __shared__ unsigned short Vs[2][64][64];     // [buf][key][d]
    __shared__ unsigned short Ps[8][16][64];     // per-wave P tile

    const int tid   = threadIdx.x;
    const int wave  = tid >> 5;
    const int lane  = tid & 31;
    const int mlan  = lane & 15;
    const int hlan  = lane >> 4;

    const int qtile = blockIdx.x;
    const int bh    = blockIdx.y;
    const int b     = bh >> 4;
    const int head  = bh & 15;
    const size_t basebh = (size_t)bh * T_SZ * HS_SZ;

    const int qbase = qtile * 128 + wave * 16;

    // Q fragments (A-layout, 16x32 each for d in [0,32) and [32,64))
    Frag qf[2];
#pragma unroll
    for (int kb = 0; kb < 2; ++kb)
#pragma unroll
        for (int v = 0; v < 8; ++v) {
            int d = kb * 32 + ((v >> 2) << 4) + hlan * 8 + ((v & 3) << 1);
            qf[kb].u[v] =
                *(const unsigned int*)&qh[basebh + (size_t)(qbase + mlan) * HS_SZ + d];
        }

    v8f oacc[4] = {};
    float mrow[8], lrow[8];
#pragma unroll
    for (int r = 0; r < 8; ++r) { mrow[r] = -1e30f; lrow[r] = 0.0f; }

    const int ntiles = 2 * qtile + 2;           // keys 0 .. qtile*128+127

    // Prologue: DMA tile 0 into buffer 0.
    if (wave == 0) {
        tdm_load_tile64x64_bf16((unsigned)(uintptr_t)&Ks[0][0][0], &kh[basebh]);
        tdm_load_tile64x64_bf16((unsigned)(uintptr_t)&Vs[0][0][0], &vh[basebh]);
    }

    for (int kt = 0; kt < ntiles; ++kt) {
        const int krow0 = kt * 64;
        const int cur = kt & 1;

        if (wave == 0) __builtin_amdgcn_s_wait_tensorcnt(0);  // tile kt landed
        __syncthreads();  // all waves: tile kt visible, buffer cur^1 free

        // Kick off DMA for tile kt+1 into the other buffer; overlaps compute.
        if (wave == 0 && kt + 1 < ntiles) {
            size_t off = basebh + (size_t)(kt + 1) * 64 * HS_SZ;
            tdm_load_tile64x64_bf16((unsigned)(uintptr_t)&Ks[cur ^ 1][0][0], &kh[off]);
            tdm_load_tile64x64_bf16((unsigned)(uintptr_t)&Vs[cur ^ 1][0][0], &vh[off]);
        }

        // S = Q K^T (contraction over d), 4 column tiles of 16 keys
        v8f sacc[4];
#pragma unroll
        for (int j = 0; j < 4; ++j) {
            v8f z = {};
#pragma unroll
            for (int db = 0; db < 2; ++db) {
                Frag kf;
#pragma unroll
                for (int v = 0; v < 8; ++v) {
                    int d = db * 32 + ((v >> 2) << 4) + hlan * 8 + ((v & 3) << 1);
                    kf.u[v] = *(const unsigned int*)&Ks[cur][j * 16 + mlan][d];
                }
                z = wmma_bf16(qf[db].v, kf.v, z);
            }
            sacc[j] = z;
        }

        // causal mask (C-layout: lane col = mlan, row = r + 8*hlan)
#pragma unroll
        for (int j = 0; j < 4; ++j) {
            int key = krow0 + j * 16 + mlan;
#pragma unroll
            for (int r = 0; r < 8; ++r) {
                int qrow = qbase + r + 8 * hlan;
                if (key > qrow) sacc[j][r] = -1e30f;
            }
        }

        // online softmax per row
#pragma unroll
        for (int r = 0; r < 8; ++r) {
            float mx = -1e30f;
#pragma unroll
            for (int j = 0; j < 4; ++j) mx = fmaxf(mx, sacc[j][r]);
#pragma unroll
            for (int off = 8; off >= 1; off >>= 1)
                mx = fmaxf(mx, __shfl_xor(mx, off, 32));
            float mnew = fmaxf(mrow[r], mx);
            float scale = __expf(mrow[r] - mnew);
            mrow[r] = mnew;
            float rsum = 0.0f;
#pragma unroll
            for (int j = 0; j < 4; ++j) {
                float p = __expf(sacc[j][r] - mnew);
                sacc[j][r] = p;
                rsum += p;
            }
#pragma unroll
            for (int off = 8; off >= 1; off >>= 1)
                rsum += __shfl_xor(rsum, off, 32);
            lrow[r] = lrow[r] * scale + rsum;
#pragma unroll
            for (int jd = 0; jd < 4; ++jd) oacc[jd][r] *= scale;
            // store P row to per-wave LDS (bf16)
            int prow = r + 8 * hlan;
#pragma unroll
            for (int j = 0; j < 4; ++j)
                Ps[wave][prow][j * 16 + mlan] = f2bf(sacc[j][r]);
        }

        // O += P V  (contraction over 64 keys -> 2 A fragments).
        // V consumed row-major: B-fragment dwords packed from 16-bit gathers.
        Frag pf[2];
#pragma unroll
        for (int kb = 0; kb < 2; ++kb)
#pragma unroll
            for (int v = 0; v < 8; ++v) {
                int kk = kb * 32 + ((v >> 2) << 4) + hlan * 8 + ((v & 3) << 1);
                pf[kb].u[v] = *(const unsigned int*)&Ps[wave][mlan][kk];
            }
#pragma unroll
        for (int jd = 0; jd < 4; ++jd) {
            int d = jd * 16 + mlan;
#pragma unroll
            for (int kb = 0; kb < 2; ++kb) {
                Frag vf;
#pragma unroll
                for (int v = 0; v < 8; ++v) {
                    int kk = kb * 32 + ((v >> 2) << 4) + hlan * 8 + ((v & 3) << 1);
                    unsigned lo = Vs[cur][kk][d];
                    unsigned hi = Vs[cur][kk + 1][d];
                    vf.u[v] = lo | (hi << 16);
                }
                oacc[jd] = wmma_bf16(pf[kb].v, vf.v, oacc[jd]);
            }
        }
    }

    // normalize + write [B,T,C] bf16
#pragma unroll
    for (int r = 0; r < 8; ++r) {
        float inv = 1.0f / lrow[r];
        int trow = qbase + r + 8 * hlan;
#pragma unroll
        for (int jd = 0; jd < 4; ++jd) {
            int d = jd * 16 + mlan;
            attnb[((size_t)(b * T_SZ + trow)) * C_SZ + head * HS_SZ + d] =
                f2bf(oacc[jd][r] * inv);
        }
    }
}

// ---------------------------------------------------------------------------
// Output projection: [8192,1024] x [1024,1024] + bias -> fp32 d_out
// ---------------------------------------------------------------------------
__global__ __launch_bounds__(256) void proj_gemm(
    const unsigned short* __restrict__ A,     // attn out, bf16 [M,C]
    const unsigned short* __restrict__ Bm,    // W_proj, bf16 [C,C]
    const float* __restrict__ bias,
    float* __restrict__ out) {
    __shared__ unsigned short As[128][32];
    __shared__ unsigned short Bs[128][32];

    const int tid   = threadIdx.x;
    const int wave  = tid >> 5;
    const int lane  = tid & 31;
    const int mlan  = lane & 15;
    const int hlan  = lane >> 4;
    const int waveM = wave >> 2;
    const int waveN = wave & 3;

    const int rowBlk = blockIdx.y * 128;
    const int colBlk = blockIdx.x * 128;

    v8f acc[4][2] = {};

    const int arow = tid >> 2;
    const int acol = (tid & 3) * 8;
    const int brow = tid >> 4;
    const int bcol = (tid & 15) * 8;

    for (int k0 = 0; k0 < C_SZ; k0 += 32) {
        __syncthreads();
#pragma unroll
        for (int p = 0; p < 2; ++p) {
            int r = p * 64 + arow;
            *(uint4*)&As[r][acol] =
                *(const uint4*)&A[(size_t)(rowBlk + r) * C_SZ + k0 + acol];
        }
#pragma unroll
        for (int p = 0; p < 2; ++p) {
            int kk = p * 16 + brow;
            uint4 t = *(const uint4*)&Bm[(size_t)(k0 + kk) * C_SZ + colBlk + bcol];
            const unsigned short* e = (const unsigned short*)&t;
#pragma unroll
            for (int i = 0; i < 8; ++i) Bs[bcol + i][kk] = e[i];
        }
        if (k0 + 32 < C_SZ) {
            __builtin_prefetch(&A[(size_t)(rowBlk + arow) * C_SZ + k0 + 32 + acol], 0, 3);
            __builtin_prefetch(&Bm[(size_t)(k0 + 32 + brow) * C_SZ + colBlk + bcol], 0, 3);
        }
        __syncthreads();

        Frag af[4];
#pragma unroll
        for (int mt = 0; mt < 4; ++mt) {
            int rr = waveM * 64 + mt * 16 + mlan;
#pragma unroll
            for (int v = 0; v < 8; ++v) {
                int kk = ((v >> 2) << 4) + hlan * 8 + ((v & 3) << 1);
                af[mt].u[v] = *(const unsigned int*)&As[rr][kk];
            }
        }
        Frag bf2[2];
#pragma unroll
        for (int nt = 0; nt < 2; ++nt) {
            int cc = waveN * 32 + nt * 16 + mlan;
#pragma unroll
            for (int v = 0; v < 8; ++v) {
                int kk = ((v >> 2) << 4) + hlan * 8 + ((v & 3) << 1);
                bf2[nt].u[v] = *(const unsigned int*)&Bs[cc][kk];
            }
        }
#pragma unroll
        for (int mt = 0; mt < 4; ++mt)
#pragma unroll
            for (int nt = 0; nt < 2; ++nt)
                acc[mt][nt] = wmma_bf16(af[mt].v, bf2[nt].v, acc[mt][nt]);
    }

#pragma unroll
    for (int mt = 0; mt < 4; ++mt) {
#pragma unroll
        for (int nt = 0; nt < 2; ++nt) {
#pragma unroll
            for (int r = 0; r < 8; ++r) {
                int row = rowBlk + waveM * 64 + mt * 16 + r + 8 * hlan;
                int col = colBlk + waveN * 32 + nt * 16 + mlan;
                out[(size_t)row * C_SZ + col] = acc[mt][nt][r] + bias[col];
            }
        }
    }
}

// ---------------------------------------------------------------------------
// Host launcher. Workspace layout (needs 88 MiB):
//   [0,16M)   x bf16          [16M,22M) W_qkv bf16   [22M,24M) W_proj bf16
//   [24M,40M) Q [B,H,T,HS]    [40M,56M) K            [56M,72M) V
//   [72M,88M) attn out [B,T,C] bf16
// ---------------------------------------------------------------------------
extern "C" void kernel_launch(void* const* d_in, const int* in_sizes, int n_in,
                              void* d_out, int out_size, void* d_ws, size_t ws_size,
                              hipStream_t stream) {
    const float* x     = (const float*)d_in[0];
    const float* Wqkv  = (const float*)d_in[1];
    const float* Wproj = (const float*)d_in[2];
    const float* bproj = (const float*)d_in[3];
    float* out = (float*)d_out;

    char* ws = (char*)d_ws;
    const size_t MB = 1024 * 1024;
    unsigned short* xb     = (unsigned short*)(ws);
    unsigned short* wqkvb  = (unsigned short*)(ws + 16 * MB);
    unsigned short* wprojb = (unsigned short*)(ws + 22 * MB);
    unsigned short* qh     = (unsigned short*)(ws + 24 * MB);
    unsigned short* kh     = (unsigned short*)(ws + 40 * MB);
    unsigned short* vh     = (unsigned short*)(ws + 56 * MB);
    unsigned short* attnb  = (unsigned short*)(ws + 72 * MB);

    cvt_f32_bf16<<<2048, 256, 0, stream>>>(x, xb, M_SZ * C_SZ);
    cvt_f32_bf16<<<2048, 256, 0, stream>>>(Wqkv, wqkvb, C_SZ * N_QKV);
    cvt_f32_bf16<<<1024, 256, 0, stream>>>(Wproj, wprojb, C_SZ * C_SZ);

    qkv_gemm<<<dim3(N_QKV / 128, M_SZ / 128), 256, 0, stream>>>(xb, wqkvb, qh, kh, vh);
    flash_attn<<<dim3(T_SZ / 128, B_SZ * H_SZ), 256, 0, stream>>>(qh, kh, vh, attnb);
    proj_gemm<<<dim3(C_SZ / 128, M_SZ / 128), 256, 0, stream>>>(attnb, wprojb, bproj, out);
}
